// Injector_multimodal_32272384262539
// MI455X (gfx1250) — matrix operand order repaired and checked by reading, not compiled, not run on hardware
//
#include <hip/hip_runtime.h>
#include <hip/hip_bf16.h>
#include <stdint.h>

// ---------------- problem constants ----------------
#define B_    2
#define LQ_   4096
#define D_    768
#define NH_   6
#define NP_   4
#define DH_   128
#define HF_   128
#define WF_   128
#define NIN_  (HF_*WF_)      // 16384
#define NMOD_ 4
#define KDIM_ 768

typedef __attribute__((ext_vector_type(16))) _Float16 v16h;
typedef __attribute__((ext_vector_type(8)))  float    v8f;

union Frag { uint4 q[2]; v16h v; };
union H4   { uint2 q; _Float16 h[4]; };

// ---------------------------------------------------------------
// (NMOD, D, C) f32  ->  (NMOD, C, D) f16   (transpose + convert)
// With C == D this transposes the square projection weights so the
// GEMM can stage B tiles into LDS with b128 stores (no scalar
// transposed ds_store_b16 in the hot loop).
// ---------------------------------------------------------------
__global__ void tcvt_kernel(const float* __restrict__ src,
                            _Float16* __restrict__ dst, int C, int n) {
    int i = blockIdx.x * blockDim.x + threadIdx.x;
    if (i >= n) return;
    int m   = i / (D_ * C);
    int rem = i - m * D_ * C;
    int k   = rem / C;
    int c   = rem - k * C;
    dst[((size_t)m * C + c) * D_ + k] = (_Float16)src[i];
}

// ---------------------------------------------------------------
// LayerNorm (row of 768) -> f16. One block (256 thr) per row.
// ---------------------------------------------------------------
__global__ __launch_bounds__(256)
void ln_f16_kernel(const float* __restrict__ x,
                   const float* __restrict__ sc,
                   const float* __restrict__ bi,
                   _Float16* __restrict__ y) {
    __shared__ float s1[256], s2[256];
    const int row = blockIdx.x, tid = threadIdx.x;
    const size_t base = (size_t)row * D_;
    float v0 = x[base + tid];
    float v1 = x[base + tid + 256];
    float v2 = x[base + tid + 512];
    s1[tid] = v0 + v1 + v2;
    s2[tid] = v0 * v0 + v1 * v1 + v2 * v2;
    __syncthreads();
    for (int o = 128; o > 0; o >>= 1) {
        if (tid < o) { s1[tid] += s1[tid + o]; s2[tid] += s2[tid + o]; }
        __syncthreads();
    }
    const float mean = s1[0] * (1.0f / (float)D_);
    const float var  = s2[0] * (1.0f / (float)D_) - mean * mean;
    const float r    = rsqrtf(var + 1e-6f);
    y[base + tid]       = (_Float16)((v0 - mean) * r * sc[tid]       + bi[tid]);
    y[base + tid + 256] = (_Float16)((v1 - mean) * r * sc[tid + 256] + bi[tid + 256]);
    y[base + tid + 512] = (_Float16)((v2 - mean) * r * sc[tid + 512] + bi[tid + 512]);
}

// ---------------------------------------------------------------
// WMMA GEMM:  out = A[M x 768] (f16, row-major) * W^T[768 x 768]
// (f16, N-major) + bias.   Block tile 128x128x32, 8 waves, each
// wave computes 64x32 (4x2 tiles of 16x16) -> 8 WMMA per k-step
// against 12 ds_load_b128 (vs 4:8 before).
// MODE 0: value projection  -> f16 stored as (B, NH, NIN, DH)
// MODE 1: output projection -> f32 acc = beta*acc + gamma*(val+bias)
// ---------------------------------------------------------------
template <int MODE>
__global__ __launch_bounds__(256)
void wmma_gemm_kernel(const _Float16* __restrict__ A,
                      const _Float16* __restrict__ WT,   // (N, K) f16
                      const float* __restrict__ bias,
                      _Float16* __restrict__ outH,
                      float* __restrict__ outF,
                      const float* __restrict__ gamma,
                      int mod, float beta) {
    __shared__ _Float16 sA[2][128 * 32];   // 8 KB x2
    __shared__ _Float16 sB[2][128 * 32];   // 8 KB x2, stored [n][k]

    const int tid  = threadIdx.x;
    const int lane = tid & 31;
    const int wid  = tid >> 5;
    const int wm   = wid & 1;       // 0..1  -> M offset 64*wm
    const int wn   = wid >> 1;      // 0..3  -> N offset 32*wn
    const int bM   = blockIdx.y * 128;
    const int bN   = blockIdx.x * 128;

    // global-load assignments (A and B^T are both fetched K-contiguous)
    const int gr = tid >> 1;             // row within tile (0..127)
    const int gc = (tid & 1) * 16;       // k offset within tile (0 or 16)

    uint4 a0, a1, b0, b1;
    auto gload = [&](int k0) {
        const _Float16* ap = A  + (size_t)(bM + gr) * KDIM_ + k0 + gc;
        const _Float16* bp = WT + (size_t)(bN + gr) * KDIM_ + k0 + gc;
        a0 = ((const uint4*)ap)[0];
        a1 = ((const uint4*)ap)[1];
        b0 = ((const uint4*)bp)[0];
        b1 = ((const uint4*)bp)[1];
    };
    auto sstore = [&](int buf) {
        *((uint4*)&sA[buf][gr * 32 + gc])     = a0;
        *((uint4*)&sA[buf][gr * 32 + gc + 8]) = a1;
        *((uint4*)&sB[buf][gr * 32 + gc])     = b0;
        *((uint4*)&sB[buf][gr * 32 + gc + 8]) = b1;
    };

    v8f acc[4][2];
#pragma unroll
    for (int i = 0; i < 4; ++i)
#pragma unroll
        for (int j = 0; j < 2; ++j) acc[i][j] = 0.0f;

    gload(0);
    sstore(0);
    __syncthreads();

    const int KT = KDIM_ / 32;                 // 24 k-steps
    const int mrow = lane & 15;
    const int ksA  = (lane < 16) ? 0 : 8;      // ISA A-frag lane striping
    const int ksB  = (lane < 16) ? 0 : 8;      // ISA B-frag lane striping

    for (int kt = 0; kt < KT; ++kt) {
        const int cur = kt & 1;
        if (kt + 1 < KT) gload((kt + 1) * 32);
        if (kt + 2 < KT) {
            __builtin_prefetch(A  + (size_t)(bM + gr) * KDIM_ + (kt + 2) * 32 + gc, 0, 1);
            __builtin_prefetch(WT + (size_t)(bN + gr) * KDIM_ + (kt + 2) * 32 + gc, 0, 1);
        }

        Frag af[4], bf[2];
#pragma unroll
        for (int i = 0; i < 4; ++i) {
            const _Float16* p = &sA[cur][(wm * 64 + i * 16 + mrow) * 32];
            af[i].q[0] = *(const uint4*)(p + ksA);
            af[i].q[1] = *(const uint4*)(p + ksA + 16);
        }
#pragma unroll
        for (int j = 0; j < 2; ++j) {
            const _Float16* p = &sB[cur][(wn * 32 + j * 16 + mrow) * 32];
            bf[j].q[0] = *(const uint4*)(p + ksB);
            bf[j].q[1] = *(const uint4*)(p + ksB + 16);
        }
#pragma unroll
        for (int i = 0; i < 4; ++i)
#pragma unroll
            for (int j = 0; j < 2; ++j)
                acc[i][j] = __builtin_amdgcn_wmma_f32_16x16x32_f16(
                    false, af[i].v, false, bf[j].v,
                    (short)0, acc[i][j], false, false);

        if (kt + 1 < KT) sstore(cur ^ 1);
        __syncthreads();
    }

    // ------------------ epilogue ------------------
#pragma unroll
    for (int i = 0; i < 4; ++i)
#pragma unroll
        for (int j = 0; j < 2; ++j) {
            const int nbase = bN + wn * 32 + j * 16 + (lane & 15);
            const int mbase = bM + wm * 64 + i * 16 + ((lane < 16) ? 0 : 8);
            const float bb = bias[nbase];
#pragma unroll
            for (int r = 0; r < 8; ++r) {
                const int m = mbase + r;
                const float val = acc[i][j][r] + bb;
                if (MODE == 0) {
                    // store value proj as (B, NH, NIN, DH) f16
                    const int b   = m >> 14;           // / NIN
                    const int pix = m & (NIN_ - 1);
                    const int h   = nbase >> 7;        // / DH
                    const int dh  = nbase & (DH_ - 1);
                    outH[(((size_t)(b * NH_ + h)) * NIN_ + pix) * DH_ + dh] =
                        (_Float16)val;
                } else {
                    const int b = m >> 12;             // / LQ
                    const int q = m & (LQ_ - 1);
                    const float g =
                        gamma[((size_t)b * LQ_ + q) * NMOD_ + mod];
                    const size_t oi = (size_t)m * D_ + nbase;
                    outF[oi] = beta * outF[oi] + g * val;
                }
            }
        }
}

// ---------------------------------------------------------------
// Offsets + attention weights + sampling locations.
// One block (128 thr) per (b,q) row. 72 dot products of length 768.
// ---------------------------------------------------------------
__global__ __launch_bounds__(128)
void offaw_kernel(const _Float16* __restrict__ qn,
                  const _Float16* __restrict__ WoffT,   // (48, 768)
                  const _Float16* __restrict__ WattwT,  // (24, 768)
                  const float* __restrict__ boff,
                  const float* __restrict__ battw,
                  const float* __restrict__ refp,       // (B,LQ,1,2)
                  float* __restrict__ pxb, float* __restrict__ pyb,
                  float* __restrict__ awb) {
    __shared__ _Float16 qs[D_];
    __shared__ float res[72];
    const int row = blockIdx.x;          // b*LQ + q
    const int tid = threadIdx.x;

    for (int i = tid; i < D_; i += 128) qs[i] = qn[(size_t)row * D_ + i];
    __syncthreads();

    if (tid < 72) {
        const _Float16* w = (tid < 48) ? (WoffT + (size_t)tid * D_)
                                       : (WattwT + (size_t)(tid - 48) * D_);
        float s = (tid < 48) ? boff[tid] : battw[tid - 48];
        for (int k = 0; k < D_; ++k)
            s += (float)qs[k] * (float)w[k];
        res[tid] = s;
    }
    __syncthreads();

    if (tid < NH_) {
        const int h = tid;
        // softmax over NP=4 logits
        float lg[NP_], mx = -1e30f;
        for (int p = 0; p < NP_; ++p) {
            lg[p] = res[48 + h * NP_ + p];
            mx = fmaxf(mx, lg[p]);
        }
        float den = 0.f;
        for (int p = 0; p < NP_; ++p) { lg[p] = expf(lg[p] - mx); den += lg[p]; }
        const float inv = 1.0f / den;
        const float r0 = refp[(size_t)row * 2 + 0];
        const float r1 = refp[(size_t)row * 2 + 1];
        for (int p = 0; p < NP_; ++p) {
            const float ox = res[h * (NP_ * 2) + p * 2 + 0];
            const float oy = res[h * (NP_ * 2) + p * 2 + 1];
            const size_t idx = ((size_t)row * NH_ + h) * NP_ + p;
            pxb[idx] = r0 * (float)WF_ + ox - 0.5f;   // (ref + off/W)*W - 0.5
            pyb[idx] = r1 * (float)HF_ + oy - 0.5f;
            awb[idx] = lg[p] * inv;
        }
    }
}

// ---------------------------------------------------------------
// Deformable bilinear sampling.  One wave per (b, q, h);
// lane handles 4 channels of DH=128.  v layout (B, NH, NIN, DH) f16.
// ---------------------------------------------------------------
__global__ __launch_bounds__(256)
void sample_kernel(const _Float16* __restrict__ v,
                   const float* __restrict__ pxb,
                   const float* __restrict__ pyb,
                   const float* __restrict__ awb,
                   _Float16* __restrict__ outH) {
    const int tid  = threadIdx.x;
    const int lane = tid & 31;
    const int wid  = tid >> 5;
    const int task = blockIdx.x * 8 + wid;        // b*(LQ*NH) + q*NH + h
    const int b = task / (LQ_ * NH_);
    const int r = task - b * (LQ_ * NH_);
    const int q = r / NH_;
    const int h = r - q * NH_;

    const _Float16* vb = v + ((size_t)(b * NH_ + h)) * NIN_ * DH_;
    const int c0 = lane * 4;

    float a0 = 0.f, a1 = 0.f, a2 = 0.f, a3 = 0.f;

#pragma unroll
    for (int p = 0; p < NP_; ++p) {
        const size_t pi = (size_t)task * NP_ + p;
        const float px = pxb[pi], py = pyb[pi], aw = awb[pi];
        const float x0f = floorf(px), y0f = floorf(py);
        const float fx = px - x0f, fy = py - y0f;
        const int x0 = (int)x0f, y0 = (int)y0f;
        const float w00 = aw * (1.f - fy) * (1.f - fx);
        const float w01 = aw * (1.f - fy) * fx;
        const float w10 = aw * fy * (1.f - fx);
        const float w11 = aw * fy * fx;

        const int ys[2] = { y0, y0 + 1 };
        const int xs[2] = { x0, x0 + 1 };
        const float ww[4] = { w00, w01, w10, w11 };
#pragma unroll
        for (int cy = 0; cy < 2; ++cy)
#pragma unroll
            for (int cx = 0; cx < 2; ++cx) {
                const int yi = ys[cy], xi = xs[cx];
                const float w = ww[cy * 2 + cx];
                if (yi >= 0 && yi < HF_ && xi >= 0 && xi < WF_) {
                    H4 d;
                    d.q = *(const uint2*)(vb + ((size_t)(yi * WF_ + xi)) * DH_ + c0);
                    a0 += w * (float)d.h[0];
                    a1 += w * (float)d.h[1];
                    a2 += w * (float)d.h[2];
                    a3 += w * (float)d.h[3];
                }
            }
    }

    H4 o;
    o.h[0] = (_Float16)a0; o.h[1] = (_Float16)a1;
    o.h[2] = (_Float16)a2; o.h[3] = (_Float16)a3;
    *(uint2*)(outH + ((size_t)(b * LQ_ + q)) * D_ + h * DH_ + c0) = o.q;
}

// ---------------------------------------------------------------
// Final residual add:  out = query + acc
// ---------------------------------------------------------------
__global__ void final_add_kernel(const float* __restrict__ query,
                                 const float* __restrict__ acc,
                                 float* __restrict__ out, int n) {
    int i = blockIdx.x * blockDim.x + threadIdx.x;
    if (i < n) out[i] = query[i] + acc[i];
}

// ---------------------------------------------------------------
extern "C" void kernel_launch(void* const* d_in, const int* in_sizes, int n_in,
                              void* d_out, int out_size, void* d_ws, size_t ws_size,
                              hipStream_t stream) {
    (void)in_sizes; (void)n_in; (void)ws_size;

    const float* query = (const float*)d_in[0];
    const float* refp  = (const float*)d_in[1];
    const float* feat  = (const float*)d_in[2];
    const float* feato = (const float*)d_in[3];
    const float* gamma = (const float*)d_in[6];
    const float* qns   = (const float*)d_in[7];
    const float* qnb   = (const float*)d_in[8];
    const float* fns   = (const float*)d_in[9];
    const float* fnb   = (const float*)d_in[10];
    const float* Wv    = (const float*)d_in[11];
    const float* bv    = (const float*)d_in[12];
    const float* Woff  = (const float*)d_in[13];
    const float* boff  = (const float*)d_in[14];
    const float* Wattw = (const float*)d_in[15];
    const float* battw = (const float*)d_in[16];
    const float* Wout  = (const float*)d_in[17];
    const float* bout  = (const float*)d_in[18];
    float* out = (float*)d_out;

    // ---------------- workspace layout ----------------
    char* ws = (char*)d_ws;
    size_t off = 0;
    auto alloc = [&](size_t bytes) -> char* {
        off = (off + 255) & ~(size_t)255;
        char* p = ws + off;
        off += bytes;
        return p;
    };
    const size_t MQ = (size_t)B_ * LQ_;     // 8192
    const size_t MF = (size_t)B_ * NIN_;    // 32768

    _Float16* qnH    = (_Float16*)alloc(MQ * D_ * 2);
    _Float16* WvT    = (_Float16*)alloc((size_t)NMOD_ * D_ * D_ * 2);
    _Float16* WoutT  = (_Float16*)alloc((size_t)NMOD_ * D_ * D_ * 2);
    _Float16* WoffT  = (_Float16*)alloc((size_t)NMOD_ * 48 * D_ * 2);
    _Float16* WattwT = (_Float16*)alloc((size_t)NMOD_ * 24 * D_ * 2);
    _Float16* fnH    = (_Float16*)alloc(MF * D_ * 2);
    _Float16* vH     = (_Float16*)alloc((size_t)B_ * NH_ * NIN_ * DH_ * 2);
    float*    pxb    = (float*)alloc(MQ * NH_ * NP_ * 4);
    float*    pyb    = (float*)alloc(MQ * NH_ * NP_ * 4);
    float*    awb    = (float*)alloc(MQ * NH_ * NP_ * 4);
    _Float16* msdaH  = (_Float16*)alloc(MQ * D_ * 2);
    float*    accF   = (float*)alloc(MQ * D_ * 4);

    // ---------------- one-time weight prep (transpose + f16) ----------------
    {
        int n = NMOD_ * D_ * D_;
        tcvt_kernel<<<(n + 255) / 256, 256, 0, stream>>>(Wv, WvT, D_, n);
        tcvt_kernel<<<(n + 255) / 256, 256, 0, stream>>>(Wout, WoutT, D_, n);
        int n1 = NMOD_ * D_ * 48;
        tcvt_kernel<<<(n1 + 255) / 256, 256, 0, stream>>>(Woff, WoffT, 48, n1);
        int n2 = NMOD_ * D_ * 24;
        tcvt_kernel<<<(n2 + 255) / 256, 256, 0, stream>>>(Wattw, WattwT, 24, n2);
    }

    // ---------------- query layernorm ----------------
    ln_f16_kernel<<<(int)MQ, 256, 0, stream>>>(query, qns, qnb, qnH);

    // ---------------- per-modality pipeline ----------------
    for (int m = 0; m < NMOD_; ++m) {
        const float* fsrc =
            (m == 0) ? feat : (feato + (size_t)(m - 1) * B_ * NIN_ * D_);

        // 1) feat layernorm -> f16
        ln_f16_kernel<<<(int)MF, 256, 0, stream>>>(
            fsrc, fns + (size_t)m * D_, fnb + (size_t)m * D_, fnH);

        // 2) value projection (WMMA) -> (B,NH,NIN,DH) f16
        wmma_gemm_kernel<0><<<dim3(D_ / 128, (int)(MF / 128)), 256, 0, stream>>>(
            fnH, WvT + (size_t)m * D_ * D_, bv + (size_t)m * D_,
            vH, nullptr, nullptr, m, 0.0f);

        // 3) offsets / attention weights / sampling locations
        offaw_kernel<<<(int)MQ, 128, 0, stream>>>(
            qnH, WoffT + (size_t)m * 48 * D_, WattwT + (size_t)m * 24 * D_,
            boff + (size_t)m * 48, battw + (size_t)m * 24, refp,
            pxb, pyb, awb);

        // 4) deformable bilinear sampling -> (B,LQ,D) f16
        sample_kernel<<<(int)(B_ * LQ_ * NH_ / 8), 256, 0, stream>>>(
            vH, pxb, pyb, awb, msdaH);

        // 5) output projection (WMMA) with gamma-weighted accumulate
        wmma_gemm_kernel<1><<<dim3(D_ / 128, (int)(MQ / 128)), 256, 0, stream>>>(
            msdaH, WoutT + (size_t)m * D_ * D_, bout + (size_t)m * D_,
            nullptr, accF, gamma, m, (m == 0) ? 0.0f : 1.0f);
    }

    // ---------------- residual ----------------
    final_add_kernel<<<(out_size + 255) / 256, 256, 0, stream>>>(
        query, accF, out, out_size);
}
